// SimpleCNN_10617159156444
// MI455X (gfx1250) — matrix operand order repaired and checked by reading, not compiled, not run on hardware
//
#include <hip/hip_runtime.h>
#include <math.h>

// ---------------------------------------------------------------------------
// CDNA5 (gfx1250) implementation of the clustered-conv SimpleCNN reference.
// Distance / soft-assignment GEMMs -> v_wmma_f32_16x16x32_f16 (wave32).
// Centers are staged into LDS with GLOBAL_LOAD_ASYNC_TO_LDS_B128 (ASYNCcnt)
// when the builtins are available, then consumed by all 8 waves of the block.
// WMMA C/D layout: lane -> N column, VGPR r -> M = r + 8*(lane/16).
// ---------------------------------------------------------------------------

#define KTEMP 1e-5f
#define MAXDP 160          // max padded point dim (stage2: 144 -> 160)
#define CCHUNK 64          // centers staged per LDS chunk (64*160*2B = 20KB)

typedef _Float16 v16h __attribute__((ext_vector_type(16)));
typedef float    v8f  __attribute__((ext_vector_type(8)));
typedef int      v4i  __attribute__((ext_vector_type(4)));

#define AS1 __attribute__((address_space(1)))
#define AS3 __attribute__((address_space(3)))

union U16H { v16h v; _Float16 e[16]; };
union U8F  { v8f  v; float    e[8];  };

#if __has_builtin(__builtin_amdgcn_global_load_async_to_lds_b128)
#define HAVE_ASYNC_COPY 1
#endif

#ifdef HAVE_ASYNC_COPY
static __device__ __forceinline__ void wait_async0() {
#if __has_builtin(__builtin_amdgcn_s_wait_asynccnt)
  __builtin_amdgcn_s_wait_asynccnt(0);
#else
  asm volatile("s_wait_asynccnt 0" ::: "memory");
#endif
}
#endif

// A fragment (16x32 f16, MxK): lane L holds row m=L%16; VGPR vi holds K pair at
// (L>=16 ? 16:0) + vi*2   (contiguous k order, matching the B fragments)
static __device__ __forceinline__ v16h a_frag(const _Float16* __restrict__ rowptr,
                                              int kc, int half) {
  U16H a;
#pragma unroll
  for (int vi = 0; vi < 8; ++vi) {
    int k = kc + (half ? 16 : 0) + 2 * vi;
    a.e[2 * vi]     = rowptr[k];
    a.e[2 * vi + 1] = rowptr[k + 1];
  }
  return a.v;
}

// B fragment (32x16 f16, KxN) where B[k][n] = row-major source row n, element k
// (i.e. B = srcRow^T): pairs contiguous along k within the row.
static __device__ __forceinline__ v16h b_frag_rowmajor(const _Float16* __restrict__ rowptr,
                                                       int kc, int half) {
  U16H b;
#pragma unroll
  for (int vi = 0; vi < 8; ++vi) {
    int k = kc + (half ? 16 : 0) + 2 * vi;
    b.e[2 * vi]     = rowptr[k];
    b.e[2 * vi + 1] = rowptr[k + 1];
  }
  return b.v;
}

// B fragment where B[k][n] = src[k][n] (k strided by `stride` elements).
static __device__ __forceinline__ v16h b_frag_colmajor(const _Float16* __restrict__ colptr,
                                                       int kc, int half, int stride) {
  U16H b;
#pragma unroll
  for (int vi = 0; vi < 8; ++vi) {
    int k = kc + (half ? 16 : 0) + 2 * vi;
    b.e[2 * vi]     = colptr[(size_t)k * stride];
    b.e[2 * vi + 1] = colptr[(size_t)(k + 1) * stride];
  }
  return b.v;
}

// ------------------------------ data prep ----------------------------------

// F.unfold(k=3,pad=1,stride=1): P[b][(c*9+kh*3+kw)][y*W+x]
__global__ void k_unfold(const float* __restrict__ x, float* __restrict__ P,
                         int B, int C, int H, int W) {
  int total = B * C * 9 * H * W;
  for (int i = blockIdx.x * blockDim.x + threadIdx.x; i < total;
       i += gridDim.x * blockDim.x) {
    int t = i;
    int xw = t % W; t /= W;
    int yh = t % H; t /= H;
    int kw = t % 3; t /= 3;
    int kh = t % 3; t /= 3;
    int c  = t % C; t /= C;
    int b  = t;
    int iy = yh + kh - 1, ix = xw + kw - 1;
    float v = 0.f;
    if (iy >= 0 && iy < H && ix >= 0 && ix < W)
      v = x[((size_t)(b * C + c) * H + iy) * W + ix];
    P[i] = v;
  }
}

// f16 padded copy of the [N x D] point matrix (pad cols D..Dp-1 with zero)
__global__ void k_pack_points(const float* __restrict__ P, _Float16* __restrict__ pH,
                              int N, int D, int Dp) {
  size_t total = (size_t)N * Dp;
  for (size_t i = blockIdx.x * (size_t)blockDim.x + threadIdx.x; i < total;
       i += gridDim.x * (size_t)blockDim.x) {
    int n = (int)(i / Dp), k = (int)(i % Dp);
    pH[i] = (k < D) ? (_Float16)P[(size_t)n * D + k] : (_Float16)0.f;
  }
}

__global__ void k_pack_centers(const float* __restrict__ centers,
                               _Float16* __restrict__ cH, int D, int Dp) {
  int total = 256 * Dp;
  for (int i = blockIdx.x * blockDim.x + threadIdx.x; i < total;
       i += gridDim.x * blockDim.x) {
    int j = i / Dp, k = i % Dp;
    cH[i] = (k < D) ? (_Float16)centers[j * D + k] : (_Float16)0.f;
  }
}

__global__ void k_rownorm(const float* __restrict__ M, float* __restrict__ out,
                          int rows, int D) {
  for (int n = blockIdx.x * blockDim.x + threadIdx.x; n < rows;
       n += gridDim.x * blockDim.x) {
    float s = 0.f;
    for (int k = 0; k < D; ++k) { float v = M[(size_t)n * D + k]; s += v * v; }
    out[n] = s;
  }
}

// deterministic pseudo-permutation row pick (stand-in for jax.random.permutation)
__global__ void k_init_centers(const float* __restrict__ P, float* __restrict__ centers,
                               int N, int D) {
  int total = 256 * D;
  for (int i = blockIdx.x * blockDim.x + threadIdx.x; i < total;
       i += gridDim.x * blockDim.x) {
    int j = i / D, k = i % D;
    unsigned row = (unsigned)(((unsigned long long)j * 2654435761ull) % (unsigned)N);
    centers[i] = P[(size_t)row * D + k];
  }
}

__global__ void k_zero(float* __restrict__ p, int n) {
  for (int i = blockIdx.x * blockDim.x + threadIdx.x; i < n;
       i += gridDim.x * blockDim.x)
    p[i] = 0.f;
}

// ------------------------------ k-means ------------------------------------

// labels[n] = argmin_j || flat[n] - centers[j] ||   via WMMA dot products.
// Block = 8 waves; each wave owns a 16-point tile. Centers are staged into LDS
// in chunks of CCHUNK rows (async copy when available) and shared by all waves.
__global__ void k_assign(const _Float16* __restrict__ pH,
                         const _Float16* __restrict__ cH,
                         const float* __restrict__ anorm,
                         const float* __restrict__ bnorm,
                         int* __restrict__ labels, int N, int Dp) {
  __shared__ __align__(16) _Float16 ldsC[CCHUNK * MAXDP];

  int tid  = threadIdx.x;
  int wave = (blockIdx.x * blockDim.x + tid) >> 5;
  int lane = tid & 31;
  int m0 = wave * 16;
  bool active = (m0 < N);
  int half = lane >> 4, l16 = lane & 15;

  float anr[8];
  float bestv[8]; int besti[8];
#pragma unroll
  for (int r = 0; r < 8; ++r) {
    anr[r] = active ? anorm[m0 + r + 8 * half] : 0.f;
    bestv[r] = 3.4e38f; besti[r] = 0;
  }

  const _Float16* arow = active ? &pH[(size_t)(m0 + l16) * Dp] : pH;

  const int units16 = (CCHUNK * Dp) / 8;  // 16-byte packets per chunk
  for (int cc = 0; cc < 256; cc += CCHUNK) {
    // ---- stage CCHUNK center rows into LDS (all 256 threads) ----
    const _Float16* src = &cH[(size_t)cc * Dp];
#ifdef HAVE_ASYNC_COPY
    for (int u = tid; u < units16; u += blockDim.x) {
      __builtin_amdgcn_global_load_async_to_lds_b128(
          (AS1 v4i*)(void*)&src[u * 8],
          (AS3 v4i*)(void*)&ldsC[u * 8],
          0, 0);
    }
    wait_async0();
#else
    for (int u = tid; u < units16 * 8; u += blockDim.x) ldsC[u] = src[u];
#endif
    __syncthreads();

    if (active) {
      for (int n0 = 0; n0 < CCHUNK; n0 += 16) {
        U8F acc;
#pragma unroll
        for (int r = 0; r < 8; ++r) acc.e[r] = 0.f;
        const _Float16* brow = &ldsC[(n0 + l16) * Dp];
        for (int kc = 0; kc < Dp; kc += 32) {
          if (kc + 32 < Dp) __builtin_prefetch(&arow[kc + 32], 0, 0);  // global_prefetch_b8
          v16h a = a_frag(arow, kc, half);
          v16h b = b_frag_rowmajor(brow, kc, half);
          acc.v = __builtin_amdgcn_wmma_f32_16x16x32_f16(
              false, a, false, b, (short)0, acc.v, false, false);
        }
        int n = cc + n0 + l16;
        float bn = bnorm[n];
#pragma unroll
        for (int r = 0; r < 8; ++r) {
          float d2 = anr[r] + bn - 2.f * acc.e[r];
          if (d2 < bestv[r]) { bestv[r] = d2; besti[r] = n; }
        }
      }
    }
    __syncthreads();
  }

  if (active) {
    // min-reduce across the 16 lanes of each half (first-index tiebreak)
#pragma unroll
    for (int msk = 1; msk < 16; msk <<= 1) {
#pragma unroll
      for (int r = 0; r < 8; ++r) {
        float ov = __shfl_xor(bestv[r], msk, 32);
        int   oi = __shfl_xor(besti[r], msk, 32);
        if (ov < bestv[r] || (ov == bestv[r] && oi < besti[r])) {
          bestv[r] = ov; besti[r] = oi;
        }
      }
    }
    if (l16 == 0) {
#pragma unroll
      for (int r = 0; r < 8; ++r) labels[m0 + r + 8 * half] = besti[r];
    }
  }
}

__global__ void k_accum(const float* __restrict__ P, const int* __restrict__ labels,
                        float* __restrict__ sums, float* __restrict__ counts,
                        int N, int D) {
  for (int n = blockIdx.x * blockDim.x + threadIdx.x; n < N;
       n += gridDim.x * blockDim.x) {
    int l = labels[n];
    atomicAdd(&counts[l], 1.f);
    for (int k = 0; k < D; ++k)
      atomicAdd(&sums[l * D + k], P[(size_t)n * D + k]);
  }
}

__global__ void k_update(float* __restrict__ centers, const float* __restrict__ sums,
                         const float* __restrict__ counts, int D) {
  int total = 256 * D;
  for (int i = blockIdx.x * blockDim.x + threadIdx.x; i < total;
       i += gridDim.x * blockDim.x) {
    float c = counts[i / D];
    if (c > 0.f) centers[i] = sums[i] / fmaxf(c, 1.f);
  }
}

// ------------------- soft assignment + convex blend ------------------------

// One wave (32-thread block) per 16-point tile:
//   d = cdist(tile, centers) via WMMA -> LDS; softmax(-d*TEMP) -> f16 LDS;
//   tp = soft @ centers via WMMA; final = a*tp + b*flat.
__global__ void k_softblend(const _Float16* __restrict__ pH,
                            const _Float16* __restrict__ cH,
                            const float* __restrict__ P,
                            const float* __restrict__ anorm,
                            const float* __restrict__ bnorm,
                            float* __restrict__ FINAL, int N, int D, int Dp) {
  __shared__ float    sD[16 * 256];
  __shared__ _Float16 sS[16 * 256];
  int m0 = blockIdx.x * 16;
  int lane = threadIdx.x & 31;
  int half = lane >> 4, l16 = lane & 15;

  float anr[8];
#pragma unroll
  for (int r = 0; r < 8; ++r) anr[r] = anorm[m0 + r + 8 * half];

  const _Float16* arow = &pH[(size_t)(m0 + l16) * Dp];
  for (int n0 = 0; n0 < 256; n0 += 16) {
    U8F acc;
#pragma unroll
    for (int r = 0; r < 8; ++r) acc.e[r] = 0.f;
    const _Float16* brow = &cH[(size_t)(n0 + l16) * Dp];
    for (int kc = 0; kc < Dp; kc += 32) {
      v16h a = a_frag(arow, kc, half);
      v16h b = b_frag_rowmajor(brow, kc, half);
      acc.v = __builtin_amdgcn_wmma_f32_16x16x32_f16(
          false, a, false, b, (short)0, acc.v, false, false);
    }
    int n = n0 + l16;
    float bn = bnorm[n];
#pragma unroll
    for (int r = 0; r < 8; ++r) {
      float d2 = anr[r] + bn - 2.f * acc.e[r];
      sD[(r + 8 * half) * 256 + n] = sqrtf(fmaxf(d2, 1e-12f));
    }
  }
  __syncthreads();

  // stable softmax per row (all 32 lanes cooperate per row)
  for (int m = 0; m < 16; ++m) {
    float mx = -3.4e38f;
    for (int c = lane; c < 256; c += 32) mx = fmaxf(mx, -sD[m * 256 + c] * KTEMP);
#pragma unroll
    for (int s = 16; s > 0; s >>= 1) mx = fmaxf(mx, __shfl_xor(mx, s, 32));
    float ev[8]; float sum = 0.f;
#pragma unroll
    for (int t = 0; t < 8; ++t) {
      int c = lane + 32 * t;
      ev[t] = __expf(-sD[m * 256 + c] * KTEMP - mx);
      sum += ev[t];
    }
#pragma unroll
    for (int s = 16; s > 0; s >>= 1) sum += __shfl_xor(sum, s, 32);
    float inv = 1.f / sum;
#pragma unroll
    for (int t = 0; t < 8; ++t)
      sS[m * 256 + lane + 32 * t] = (_Float16)(ev[t] * inv);
  }
  __syncthreads();

  // tp = soft(16x256) @ centers(256xD); blend and store
  const float cA = KTEMP / (KTEMP + 1.f);
  const float cB = 1.f / (KTEMP + 1.f);
  const _Float16* srow = &sS[l16 * 256];
  for (int n0 = 0; n0 < D; n0 += 16) {
    U8F acc;
#pragma unroll
    for (int r = 0; r < 8; ++r) acc.e[r] = 0.f;
    const _Float16* bcol = &cH[n0 + l16];
    for (int kc = 0; kc < 256; kc += 32) {
      v16h a = a_frag(srow, kc, half);
      v16h b = b_frag_colmajor(bcol, kc, half, Dp);
      acc.v = __builtin_amdgcn_wmma_f32_16x16x32_f16(
          false, a, false, b, (short)0, acc.v, false, false);
    }
    int col = n0 + l16;
#pragma unroll
    for (int r = 0; r < 8; ++r) {
      if (col < D) {
        size_t idx = (size_t)(m0 + r + 8 * half) * D + col;
        FINAL[idx] = cA * acc.e[r] + cB * P[idx];
      }
    }
  }
}

// ----------------- fold + shift + conv(s=3) + relu + maxpool ---------------
// Algebraically the fold/shift/conv collapses to a masked dot of the per-pixel
// patch vector with the 3x3 weights (top/left border terms dropped).
__global__ void k_convpool(const float* __restrict__ FINAL, const float* __restrict__ w,
                           const float* __restrict__ bias, float* __restrict__ out,
                           int B, int C, int OC, int H, int W) {
  int Hp = H / 2, Wp = W / 2, HW = H * W;
  int total = B * OC * Hp * Wp;
  for (int i = blockIdx.x * blockDim.x + threadIdx.x; i < total;
       i += gridDim.x * blockDim.x) {
    int px = i % Wp;
    int py = (i / Wp) % Hp;
    int oc = (i / (Wp * Hp)) % OC;
    int b  = i / (Wp * Hp * OC);
    float mx = -3.4e38f;
    for (int dy = 0; dy < 2; ++dy) {
      for (int dx = 0; dx < 2; ++dx) {
        int oy = py * 2 + dy, ox = px * 2 + dx;
        float s = bias[oc];
        for (int c = 0; c < C; ++c) {
          for (int dr = 0; dr < 3; ++dr) {
            if (dr == 0 && oy == 0) continue;
            for (int dc = 0; dc < 3; ++dc) {
              if (dc == 0 && ox == 0) continue;
              s += w[((oc * C + c) * 3 + dr) * 3 + dc] *
                   FINAL[((size_t)b * C * 9 + c * 9 + dr * 3 + dc) * HW + oy * W + ox];
            }
          }
        }
        mx = fmaxf(mx, fmaxf(s, 0.f));  // relu then 2x2 maxpool
      }
    }
    out[i] = mx;
  }
}

__global__ void k_fc(const float* __restrict__ h, const float* __restrict__ fcw,
                     const float* __restrict__ fcb, float* __restrict__ out,
                     int B, int K, int O) {
  int total = B * O;
  for (int i = blockIdx.x * blockDim.x + threadIdx.x; i < total;
       i += gridDim.x * blockDim.x) {
    int o = i % O, b = i / O;
    float s = fcb[o];
    for (int k = 0; k < K; ++k) s += h[(size_t)b * K + k] * fcw[(size_t)o * K + k];
    out[i] = s;
  }
}

// ------------------------------ host side ----------------------------------

static inline int cdiv(int a, int b) { return (a + b - 1) / b; }

static void run_stage(const float* x, int B, int C, int H, int W,
                      const float* w, const float* bias, int OC, float* pooled,
                      float* P, float* FINAL, _Float16* FLATH, float* ANORM,
                      int* LABELS, float* CENTERS, _Float16* CH, float* BNORM,
                      float* SUMS, float* COUNTS, hipStream_t stream) {
  const int D  = 9 * C;
  const int Dp = (D + 31) & ~31;
  const int N  = B * H * W;          // rows of the flattened patch matrix
  const int ND = N * D;

  k_unfold<<<cdiv(ND, 256), 256, 0, stream>>>(x, P, B, C, H, W);
  k_pack_points<<<cdiv(N * Dp, 256), 256, 0, stream>>>(P, FLATH, N, D, Dp);
  k_rownorm<<<cdiv(N, 256), 256, 0, stream>>>(P, ANORM, N, D);
  k_init_centers<<<cdiv(256 * D, 256), 256, 0, stream>>>(P, CENTERS, N, D);

  for (int it = 0; it < 10; ++it) {
    k_pack_centers<<<cdiv(256 * Dp, 256), 256, 0, stream>>>(CENTERS, CH, D, Dp);
    k_rownorm<<<1, 256, 0, stream>>>(CENTERS, BNORM, 256, D);
    int waves = N / 16;
    k_assign<<<cdiv(waves, 8), 256, 0, stream>>>(FLATH, CH, ANORM, BNORM,
                                                 LABELS, N, Dp);
    k_zero<<<cdiv(256 * D, 256), 256, 0, stream>>>(SUMS, 256 * D);
    k_zero<<<1, 256, 0, stream>>>(COUNTS, 256);
    k_accum<<<cdiv(N, 256), 256, 0, stream>>>(P, LABELS, SUMS, COUNTS, N, D);
    k_update<<<cdiv(256 * D, 256), 256, 0, stream>>>(CENTERS, SUMS, COUNTS, D);
  }

  k_pack_centers<<<cdiv(256 * Dp, 256), 256, 0, stream>>>(CENTERS, CH, D, Dp);
  k_rownorm<<<1, 256, 0, stream>>>(CENTERS, BNORM, 256, D);
  k_softblend<<<N / 16, 32, 0, stream>>>(FLATH, CH, P, ANORM, BNORM, FINAL, N, D, Dp);
  k_convpool<<<cdiv(B * OC * (H / 2) * (W / 2), 256), 256, 0, stream>>>(
      FINAL, w, bias, pooled, B, C, OC, H, W);
}

extern "C" void kernel_launch(void* const* d_in, const int* in_sizes, int n_in,
                              void* d_out, int out_size, void* d_ws, size_t ws_size,
                              hipStream_t stream) {
  (void)in_sizes; (void)n_in; (void)out_size; (void)ws_size;
  const float* x   = (const float*)d_in[0];
  const float* w1  = (const float*)d_in[1];
  const float* b1  = (const float*)d_in[2];
  const float* w2  = (const float*)d_in[3];
  const float* b2  = (const float*)d_in[4];
  const float* fcw = (const float*)d_in[5];
  const float* fcb = (const float*)d_in[6];
  float* out = (float*)d_out;

  char* ws = (char*)d_ws;
  size_t off = 0;
  auto arena = [&](size_t bytes) -> void* {
    void* p = ws + off;
    off += (bytes + 255) & ~(size_t)255;
    return p;
  };

  // max sizes across both stages (stage2 dominates P/FINAL/FLATH)
  float*    P       = (float*)arena(sizeof(float) * 3612672);        // 25088*144
  float*    FINAL   = (float*)arena(sizeof(float) * 3612672);
  _Float16* FLATH   = (_Float16*)arena(sizeof(_Float16) * 4014080);  // 25088*160
  float*    ANORM   = (float*)arena(sizeof(float) * 100352);
  int*      LABELS  = (int*)arena(sizeof(int) * 100352);
  float*    CENTERS = (float*)arena(sizeof(float) * 256 * 144);
  _Float16* CH      = (_Float16*)arena(sizeof(_Float16) * 256 * 160);
  float*    BNORM   = (float*)arena(sizeof(float) * 256);
  float*    SUMS    = (float*)arena(sizeof(float) * 256 * 144);
  float*    COUNTS  = (float*)arena(sizeof(float) * 256);
  float*    H1      = (float*)arena(sizeof(float) * 128 * 16 * 14 * 14);
  float*    H2      = (float*)arena(sizeof(float) * 128 * 32 * 7 * 7);

  // stage 1: [128,1,28,28] -> relu/pool -> H1 [128,16,14,14]
  run_stage(x, 128, 1, 28, 28, w1, b1, 16, H1,
            P, FINAL, FLATH, ANORM, LABELS, CENTERS, CH, BNORM, SUMS, COUNTS, stream);
  // stage 2: H1 -> relu/pool -> H2 [128,32,7,7]
  run_stage(H1, 128, 16, 14, 14, w2, b2, 32, H2,
            P, FINAL, FLATH, ANORM, LABELS, CENTERS, CH, BNORM, SUMS, COUNTS, stream);
  // FC: [128,1568] @ fc_w.T + fc_b -> [128,10]
  k_fc<<<cdiv(128 * 10, 256), 256, 0, stream>>>(H2, fcw, fcb, out, 128, 1568, 10);
}
